// MultiPoolingModule_60198261621162
// MI455X (gfx1250) — matrix operand order repaired and checked by reading, not compiled, not run on hardware
//
#include <hip/hip_runtime.h>

// ---------------------------------------------------------------------------
// MultiPoolingModule for MI455X (gfx1250, wave32, WMMA)
//   x:[1e6,128] f32, batch:[1e6] i64 (sorted), B=16384 segments
//   out:[16384,3,128] f32
// Memory-bound on the single 512MB pass over x (~22us floor @23.3TB/s).
// MLP (1.6 GFLOP) done with v_wmma_f32_16x16x32_f16, f32 accumulate.
// ---------------------------------------------------------------------------

typedef __attribute__((ext_vector_type(16))) _Float16 v16h;
typedef __attribute__((ext_vector_type(8)))  float    v8f;

#define FDIM 128
#define BSEG 16384
#define LN_EPS 1e-5f
// dynamic LDS: W11h(16384B) + W12h(16384B) + params(5888B)
//            + 4 waves * (2080 f32 = 8320B) + 4 waves * (3072 h = 6144B)
#define LDS_BYTES (16384 + 16384 + 5888 + 4*8320 + 4*6144)

union V16H { v16h v; unsigned u[8]; };

// A-operand (16x32 f16, M x K) per ISA 7.12.2 "16-bit A-Matrix 16x32":
//   lane l (M = caller row), VGPR v:
//   K = ((v>>2)<<4) + ((l>>4)<<3) + ((v&3)<<1) + k0
__device__ __forceinline__ v16h gatherA(const _Float16* base, int rowStride,
                                        int row, int k0, int lane) {
  V16H r;
  const _Float16* q = base + row * rowStride + k0 + ((lane >> 4) << 3);
#pragma unroll
  for (int v = 0; v < 8; ++v) {
    int k = ((v >> 2) << 4) + ((v & 3) << 1);
    r.u[v] = *reinterpret_cast<const unsigned*>(q + k);
  }
  return r.v;
}

// B-operand (32x16 f16, K x N) per ISA 7.12.4 B-matrix layout:
//   same K order as A operand:
//   K = ((v>>2)<<4) + ((l>>4)<<3) + ((v&3)<<1) + k0 ;  N = caller row
__device__ __forceinline__ v16h gatherB(const _Float16* base, int rowStride,
                                        int col, int k0, int lane) {
  V16H r;
  const _Float16* q = base + col * rowStride + k0 + ((lane >> 4) << 3);
#pragma unroll
  for (int v = 0; v < 8; ++v)
    r.u[v] = *reinterpret_cast<const unsigned*>(q + ((v >> 2) << 4) + ((v & 3) << 1));
  return r.v;
}

// ---------------- 1: segment boundaries (batch is sorted) -------------------
__global__ void k_bounds(const long long* __restrict__ batch, int n,
                         int* __restrict__ seg_start) {
  int b = blockIdx.x * blockDim.x + threadIdx.x;
  if (b > BSEG) return;
  int lo = 0, hi = n;
  while (lo < hi) {
    int mid = (lo + hi) >> 1;
    if (batch[mid] < (long long)b) lo = mid + 1; else hi = mid;
  }
  seg_start[b] = lo;  // seg_start[BSEG] == n
}

// ---------------- 2: segment sum/mean/max (no atomics) ----------------------
__global__ void k_pool(const float* __restrict__ x,
                       const int* __restrict__ seg_start,
                       float* __restrict__ ssum, float* __restrict__ smean,
                       float* __restrict__ smax) {
  int b = blockIdx.x;
  int f = threadIdx.x;
  int s = seg_start[b], e = seg_start[b + 1];
  float sum = 0.0f, mx = -INFINITY;
  for (int r = s; r < e; ++r) {              // coalesced 512B per row
    float v = x[(size_t)r * FDIM + f];
    sum += v;
    mx = fmaxf(mx, v);
  }
  int cnt = e - s;
  size_t o = (size_t)b * FDIM + f;
  ssum[o]  = sum;
  smean[o] = sum / (float)(cnt > 0 ? cnt : 1);
  smax[o]  = (cnt > 0) ? mx : 0.0f;          // reference guards empty segs
}

// ---------------- 3a: batchnorm partial sums (deterministic) ----------------
__global__ void k_bnpart(const float* __restrict__ smax,
                         const float* __restrict__ smean,
                         const float* __restrict__ ssum,
                         float* __restrict__ psum, float* __restrict__ psq) {
  int chunk = blockIdx.x;          // 128 chunks of 128 rows
  int p = blockIdx.y;              // 0:max 1:mean 2:sum (stack order)
  int f = threadIdx.x;
  const float* src = (p == 0) ? smax : (p == 1) ? smean : ssum;
  float s = 0.0f, q = 0.0f;
  for (int i = 0; i < 128; ++i) {
    float v = src[(size_t)(chunk * 128 + i) * FDIM + f];
    s += v; q += v * v;
  }
  size_t o = ((size_t)p * 128 + chunk) * FDIM + f;
  psum[o] = s; psq[o] = q;
}

// ---------------- 3b: fold BN stats into per-feature affine -----------------
__global__ void k_bnfin(const float* __restrict__ psum,
                        const float* __restrict__ psq,
                        const float* __restrict__ g1, const float* __restrict__ e1,
                        const float* __restrict__ g2, const float* __restrict__ e2,
                        const float* __restrict__ g3, const float* __restrict__ e3,
                        float* __restrict__ bnS, float* __restrict__ bnB) {
  int p = blockIdx.x, f = threadIdx.x;
  float s = 0.0f, q = 0.0f;
  for (int c = 0; c < 128; ++c) {
    size_t o = ((size_t)p * 128 + c) * FDIM + f;
    s += psum[o]; q += psq[o];
  }
  const float inv = 1.0f / (float)BSEG;
  float mu  = s * inv;
  float var = q * inv - mu * mu;              // biased variance
  const float* g = (p == 0) ? g1 : (p == 1) ? g2 : g3;
  const float* e = (p == 0) ? e1 : (p == 1) ? e2 : e3;
  float scale = g[f] * rsqrtf(var + LN_EPS);
  bnS[p * FDIM + f] = scale;
  bnB[p * FDIM + f] = e[f] - mu * scale;
}

// ---------------- 4: fused BN-affine + LN + MLP(WMMA) + LN ------------------
__global__ void
__launch_bounds__(128, 1)
k_fused(const float* __restrict__ smax, const float* __restrict__ smean,
        const float* __restrict__ ssum,
        const float* __restrict__ bnS, const float* __restrict__ bnB,
        const float* __restrict__ gf1, const float* __restrict__ bf1,
        const float* __restrict__ gf11, const float* __restrict__ bf11,
        const float* __restrict__ W11, const float* __restrict__ b11,
        const float* __restrict__ W12, const float* __restrict__ b12,
        float* __restrict__ out) {
  extern __shared__ char smem[];
  _Float16* W11h = (_Float16*)smem;                 // [64][128] f16
  _Float16* W12h = W11h + 64 * FDIM;                // [128][64] f16
  float* sb11  = (float*)(W12h + FDIM * 64);        // 64
  float* sb12  = sb11 + 64;                         // 128
  float* sgf1  = sb12 + FDIM;
  float* sbf1  = sgf1 + FDIM;
  float* sgf11 = sbf1 + FDIM;
  float* sbf11 = sgf11 + FDIM;
  float* sbnS  = sbf11 + FDIM;                      // 384
  float* sbnB  = sbnS + 384;                        // 384
  float* waveF = sbnB + 384;                        // 4 * 2080 f32
  _Float16* waveH = (_Float16*)(waveF + 4 * 2080);  // 4 * 3072 f16

  const int tid = threadIdx.x, lane = tid & 31, wv = tid >> 5;
  float*    X1  = waveF + wv * 2080;                // [16][128] f32 tile
  float*    ST  = X1 + 2048;                        // 16 x {mu, rstd}
  _Float16* A16 = waveH + wv * 3072;                // [16][128] f16
  _Float16* H16 = A16 + 2048;                       // [16][64]  f16

  // stage weights/params once per block (f32 -> f16 for WMMA operands)
  for (int i = tid; i < 64 * FDIM; i += 128) W11h[i] = (_Float16)W11[i];
  for (int i = tid; i < FDIM * 64; i += 128) W12h[i] = (_Float16)W12[i];
  if (tid < 64)  sb11[tid] = b11[tid];
  if (tid < 128) {
    sb12[tid] = b12[tid];
    sgf1[tid] = gf1[tid];  sbf1[tid]  = bf1[tid];
    sgf11[tid] = gf11[tid]; sbf11[tid] = bf11[tid];
  }
  for (int i = tid; i < 384; i += 128) { sbnS[i] = bnS[i]; sbnB[i] = bnB[i]; }
  __syncthreads();

  const int tile = blockIdx.x * 4 + wv;             // one 16-row tile per wave
  const int r0 = tile * 16;                         // rows of [B*3, 128]

  // ---- load x1 tile: pooled value -> folded batchnorm affine ----
  for (int m = 0; m < 16; ++m) {
    int r = r0 + m, b = r / 3, k = r - b * 3;
    const float* src = (k == 0) ? smax : (k == 1) ? smean : ssum;
    for (int f = lane; f < FDIM; f += 32)
      X1[m * FDIM + f] = src[(size_t)b * FDIM + f] * sbnS[k * FDIM + f]
                       + sbnB[k * FDIM + f];
  }
  __syncthreads();

  // ---- LayerNorm f1 -> f16 A operand tile ----
  if (lane < 16) {
    float s = 0.0f, q = 0.0f;
    for (int f = 0; f < FDIM; ++f) { float v = X1[lane * FDIM + f]; s += v; q += v * v; }
    float mu  = s * (1.0f / FDIM);
    float var = q * (1.0f / FDIM) - mu * mu;
    ST[lane * 2] = mu; ST[lane * 2 + 1] = rsqrtf(var + LN_EPS);
  }
  __syncthreads();
  for (int i = lane; i < 16 * FDIM; i += 32) {
    int m = i >> 7, f = i & 127;
    A16[i] = (_Float16)((X1[i] - ST[m * 2]) * ST[m * 2 + 1] * sgf1[f] + sbf1[f]);
  }
  __syncthreads();

  // ---- layer 1: [16x128] @ W11^T [128x64], v_wmma_f32_16x16x32_f16 ----
  v8f z = {};
  v8f acc[4] = {z, z, z, z};
#pragma unroll
  for (int kk = 0; kk < 4; ++kk) {
    v16h a = gatherA(A16, FDIM, lane & 15, kk * 32, lane);
#pragma unroll
    for (int nt = 0; nt < 4; ++nt) {
      v16h bm = gatherB(W11h, FDIM, (lane & 15) + nt * 16, kk * 32, lane);
      acc[nt] = __builtin_amdgcn_wmma_f32_16x16x32_f16(
          false, a, false, bm, (short)0, acc[nt], false, false);
    }
  }
  // bias + ReLU, scatter D (vgpr j: M=j|j+8, lane: N) into f16 H tile
#pragma unroll
  for (int nt = 0; nt < 4; ++nt)
#pragma unroll
    for (int j = 0; j < 8; ++j) {
      int M = j + ((lane >> 4) << 3), Nn = (lane & 15) + nt * 16;
      H16[M * 64 + Nn] = (_Float16)fmaxf(acc[nt][j] + sb11[Nn], 0.0f);
    }
  __syncthreads();

  // ---- layer 2: [16x64] @ W12^T [64x128] ----
  v8f acc2[8] = {z, z, z, z, z, z, z, z};
#pragma unroll
  for (int kk = 0; kk < 2; ++kk) {
    v16h a2 = gatherA(H16, 64, lane & 15, kk * 32, lane);
#pragma unroll
    for (int nt = 0; nt < 8; ++nt) {
      v16h b2 = gatherB(W12h, 64, (lane & 15) + nt * 16, kk * 32, lane);
      acc2[nt] = __builtin_amdgcn_wmma_f32_16x16x32_f16(
          false, a2, false, b2, (short)0, acc2[nt], false, false);
    }
  }
  // bias + residual into X1 (each element owned by exactly one lane)
#pragma unroll
  for (int nt = 0; nt < 8; ++nt)
#pragma unroll
    for (int j = 0; j < 8; ++j) {
      int M = j + ((lane >> 4) << 3), Nn = (lane & 15) + nt * 16;
      X1[M * FDIM + Nn] += acc2[nt][j] + sb12[Nn];
    }
  __syncthreads();

  // ---- LayerNorm f11 + store ----
  if (lane < 16) {
    float s = 0.0f, q = 0.0f;
    for (int f = 0; f < FDIM; ++f) { float v = X1[lane * FDIM + f]; s += v; q += v * v; }
    float mu  = s * (1.0f / FDIM);
    float var = q * (1.0f / FDIM) - mu * mu;
    ST[lane * 2] = mu; ST[lane * 2 + 1] = rsqrtf(var + LN_EPS);
  }
  __syncthreads();
  for (int i = lane; i < 16 * FDIM; i += 32) {
    int m = i >> 7, f = i & 127;
    out[(size_t)(r0 + m) * FDIM + f] =
        (X1[i] - ST[m * 2]) * ST[m * 2 + 1] * sgf11[f] + sbf11[f];
  }
}

// ---------------------------------------------------------------------------
extern "C" void kernel_launch(void* const* d_in, const int* in_sizes, int n_in,
                              void* d_out, int out_size, void* d_ws, size_t ws_size,
                              hipStream_t stream) {
  (void)n_in; (void)out_size; (void)ws_size;
  const float*     x     = (const float*)d_in[0];
  // d_in[1] = edge_index (unused by the reference computation)
  const long long* batch = (const long long*)d_in[2];
  const float* g_b1 = (const float*)d_in[3],  *be_b1 = (const float*)d_in[4];
  const float* g_b2 = (const float*)d_in[5],  *be_b2 = (const float*)d_in[6];
  const float* g_b3 = (const float*)d_in[7],  *be_b3 = (const float*)d_in[8];
  const float* g_f1 = (const float*)d_in[9],  *be_f1 = (const float*)d_in[10];
  const float* g_f11= (const float*)d_in[11], *be_f11= (const float*)d_in[12];
  const float* W11  = (const float*)d_in[13], *b11   = (const float*)d_in[14];
  const float* W12  = (const float*)d_in[15], *b12   = (const float*)d_in[16];
  float* out = (float*)d_out;

  const int N = in_sizes[0] / FDIM;

  // carve workspace (~25.6 MB)
  char* ws = (char*)d_ws;
  size_t off = 0;
  auto carve = [&](size_t bytes) -> char* {
    char* p = ws + off;
    off = (off + bytes + 255) & ~(size_t)255;
    return p;
  };
  int*   seg_start = (int*)  carve((BSEG + 1) * sizeof(int));
  float* ssum  = (float*)carve((size_t)BSEG * FDIM * sizeof(float));
  float* smean = (float*)carve((size_t)BSEG * FDIM * sizeof(float));
  float* smax  = (float*)carve((size_t)BSEG * FDIM * sizeof(float));
  float* psum  = (float*)carve((size_t)3 * 128 * FDIM * sizeof(float));
  float* psq   = (float*)carve((size_t)3 * 128 * FDIM * sizeof(float));
  float* bnS   = (float*)carve(3 * FDIM * sizeof(float));
  float* bnB   = (float*)carve(3 * FDIM * sizeof(float));

  k_bounds<<<(BSEG + 1 + 255) / 256, 256, 0, stream>>>(batch, N, seg_start);
  k_pool<<<BSEG, FDIM, 0, stream>>>(x, seg_start, ssum, smean, smax);
  k_bnpart<<<dim3(128, 3), FDIM, 0, stream>>>(smax, smean, ssum, psum, psq);
  k_bnfin<<<dim3(3), FDIM, 0, stream>>>(psum, psq, g_b1, be_b1, g_b2, be_b2,
                                        g_b3, be_b3, bnS, bnB);
  // 3072 tiles of 16 rows -> 768 blocks x 4 waves (one tile per wave, uniform)
  k_fused<<<768, 128, LDS_BYTES, stream>>>(smax, smean, ssum, bnS, bnB,
                                           g_f1, be_f1, g_f11, be_f11,
                                           W11, b11, W12, b12, out);
}
